// GraphAttentionLayer_18167711662237
// MI455X (gfx1250) — matrix operand order repaired and checked by reading, not compiled, not run on hardware
//
#include <hip/hip_runtime.h>

#define N_NODES 100000
#define E_EDGES 1600000
#define IN_F    256
#define OUT_F   128
#define ALPHA_F 0.2f

typedef _Float16 v16h __attribute__((ext_vector_type(16)));
typedef _Float16 v8h  __attribute__((ext_vector_type(8)));
typedef _Float16 v4h  __attribute__((ext_vector_type(4)));
typedef float    v8f  __attribute__((ext_vector_type(8)));
typedef float    v4f  __attribute__((ext_vector_type(4)));

// ---------------------------------------------------------------------------
// Zero-init output accumulator and rowsum (ws/out are poisoned by harness).
// ---------------------------------------------------------------------------
__global__ __launch_bounds__(256) void zero_kernel(float* __restrict__ out,
                                                   float* __restrict__ rowsum,
                                                   int total_out, int n) {
  int i = blockIdx.x * 256 + threadIdx.x;
  if (i < total_out) out[i] = 0.0f;
  if (i < n) rowsum[i] = 0.0f;
}

// ---------------------------------------------------------------------------
// Pack W (256x128 f32, row-major) into per-lane WMMA B-fragments (f16).
// Fragment layout per CDNA5 ISA "16-bit B matrix":
//   lane = 16*hh + r ; b[i] = B[k0 + 8*hh + i][col], b[8+i] = B[k0 + 16 + 8*hh + i][col = ct*16 + r]
// Fragment index = (kt*8 + ct)*32 + lane, 16 contiguous halves each.
// ---------------------------------------------------------------------------
__global__ __launch_bounds__(256) void pack_w_kernel(const float* __restrict__ W,
                                                     _Float16* __restrict__ Wp) {
  int id = blockIdx.x * 256 + threadIdx.x;   // 0 .. 2047  (8 kt * 8 ct * 32 lanes)
  if (id >= 8 * 8 * 32) return;
  int lane = id & 31;
  int ct   = (id >> 5) & 7;
  int kt   = id >> 8;
  int hh = lane >> 4, r = lane & 15;
  int col = ct * 16 + r;
  int k0  = kt * 32 + 8 * hh;
  v8h lo, hi;
#pragma unroll
  for (int i = 0; i < 8; ++i) lo[i] = (_Float16)W[(k0 + i) * OUT_F + col];
#pragma unroll
  for (int i = 0; i < 8; ++i) hi[i] = (_Float16)W[(k0 + 16 + i) * OUT_F + col];
  v8h* dst = (v8h*)(Wp + (size_t)id * 16);
  dst[0] = lo;
  dst[1] = hi;
}

// ---------------------------------------------------------------------------
// h = x @ W via v_wmma_f32_16x16x32_f16.
// Block: 256 threads = 8 waves. One block computes a 16-row x 128-col tile.
// Wave w owns column tile w (16 cols). K loop: 256 / 32 = 8 WMMAs.
// x row panel (16x256) staged in LDS as f16 once per block.
// A-fragment layout per ISA: lane (hh,r): a[0..7]=A[r, k0+8hh .. +7],
//                            a[8..15]=A[r, k0+16+8hh .. +7]  (two b128 DS loads)
// D layout: acc[j] -> h[row0 + j + 8*hh][wave*16 + r]
// ---------------------------------------------------------------------------
__global__ __launch_bounds__(256) void gemm_wmma_kernel(const float* __restrict__ x,
                                                        const _Float16* __restrict__ Wp,
                                                        float* __restrict__ h) {
  __shared__ _Float16 As[16 * IN_F];  // 8 KB
  int t    = threadIdx.x;
  int row0 = blockIdx.x * 16;         // N_NODES % 16 == 0

  // Stage 16x256 f32 -> f16 into LDS (each thread converts 16 elements).
  const float* src = x + (size_t)row0 * IN_F;
#pragma unroll
  for (int i = 0; i < 4; ++i) {
    v4f v = *(const v4f*)(src + t * 16 + i * 4);
    v4h hv;
    hv.x = (_Float16)v.x; hv.y = (_Float16)v.y;
    hv.z = (_Float16)v.z; hv.w = (_Float16)v.w;
    *(v4h*)(&As[t * 16 + i * 4]) = hv;
  }
  __syncthreads();

  int wave = t >> 5, lane = t & 31;
  int hh = lane >> 4, r = lane & 15;
  v8f acc = {};
  const v16h* bfrag = (const v16h*)Wp;
#pragma unroll
  for (int kt = 0; kt < 8; ++kt) {
    int k0 = kt * 32;
    v8h alo = *(const v8h*)(&As[r * IN_F + k0 + 8 * hh]);
    v8h ahi = *(const v8h*)(&As[r * IN_F + k0 + 16 + 8 * hh]);
    v16h a = __builtin_shufflevector(alo, ahi, 0, 1, 2, 3, 4, 5, 6, 7,
                                     8, 9, 10, 11, 12, 13, 14, 15);
    v16h b = bfrag[(kt * 8 + wave) * 32 + lane];
    acc = __builtin_amdgcn_wmma_f32_16x16x32_f16(
        /*neg_a=*/false, a, /*neg_b=*/false, b,
        /*c_mod=*/(short)0, acc, /*reuse_a=*/false, /*reuse_b=*/false);
  }
  int col  = wave * 16 + r;
  float* hp = h + (size_t)(row0 + 8 * hh) * OUT_F + col;
#pragma unroll
  for (int j = 0; j < 8; ++j) hp[(size_t)j * OUT_F] = acc[j];
}

// ---------------------------------------------------------------------------
// s1[v] = h[v,:] . a1 ; s2[v] = h[v,:] . a2  (one wave per node, float4/lane)
// ---------------------------------------------------------------------------
__global__ __launch_bounds__(256) void node_score_kernel(const float* __restrict__ h,
                                                         const float* __restrict__ a,
                                                         float* __restrict__ s1,
                                                         float* __restrict__ s2) {
  int gw   = (blockIdx.x * 256 + threadIdx.x) >> 5;
  int lane = threadIdx.x & 31;
  if (gw >= N_NODES) return;
  v4f v  = ((const v4f*)(h + (size_t)gw * OUT_F))[lane];
  v4f a1 = ((const v4f*)a)[lane];
  v4f a2 = ((const v4f*)(a + OUT_F))[lane];
  float p1 = v.x * a1.x + v.y * a1.y + v.z * a1.z + v.w * a1.w;
  float p2 = v.x * a2.x + v.y * a2.y + v.z * a2.z + v.w * a2.w;
#pragma unroll
  for (int off = 16; off > 0; off >>= 1) {
    p1 += __shfl_down(p1, off, 32);
    p2 += __shfl_down(p2, off, 32);
  }
  if (lane == 0) { s1[gw] = p1; s2[gw] = p2; }
}

// ---------------------------------------------------------------------------
// Edge aggregation: one wave per edge (grid-stride).
//   w = exp(-leaky_relu(s1[src]+s2[dst]))
//   rowsum[src] += w ; out[src,:] += w * h[dst,:]  (float4 per lane, L2 atomics)
// ---------------------------------------------------------------------------
__global__ __launch_bounds__(256) void edge_agg_kernel(const int* __restrict__ edge,
                                                       const float* __restrict__ h,
                                                       const float* __restrict__ s1,
                                                       const float* __restrict__ s2,
                                                       float* __restrict__ rowsum,
                                                       float* __restrict__ out) {
  int wid  = (blockIdx.x * blockDim.x + threadIdx.x) >> 5;
  int lane = threadIdx.x & 31;
  int nw   = (gridDim.x * blockDim.x) >> 5;
  for (int e = wid; e < E_EDGES; e += nw) {
    int s = edge[e];
    int d = edge[E_EDGES + e];
    float sc = s1[s] + s2[d];
    float lr = sc > 0.0f ? sc : ALPHA_F * sc;
    float w  = __expf(-lr);
    if (lane == 0)
      __hip_atomic_fetch_add(&rowsum[s], w, __ATOMIC_RELAXED,
                             __HIP_MEMORY_SCOPE_AGENT);
    v4f v = ((const v4f*)(h + (size_t)d * OUT_F))[lane];
    float* o = out + (size_t)s * OUT_F + lane * 4;
    __hip_atomic_fetch_add(o + 0, w * v.x, __ATOMIC_RELAXED, __HIP_MEMORY_SCOPE_AGENT);
    __hip_atomic_fetch_add(o + 1, w * v.y, __ATOMIC_RELAXED, __HIP_MEMORY_SCOPE_AGENT);
    __hip_atomic_fetch_add(o + 2, w * v.z, __ATOMIC_RELAXED, __HIP_MEMORY_SCOPE_AGENT);
    __hip_atomic_fetch_add(o + 3, w * v.w, __ATOMIC_RELAXED, __HIP_MEMORY_SCOPE_AGENT);
  }
}

// ---------------------------------------------------------------------------
// out = elu(out / max(rowsum,1-if-zero))   (in place)
// ---------------------------------------------------------------------------
__global__ __launch_bounds__(256) void finalize_kernel(float* __restrict__ out,
                                                       const float* __restrict__ rowsum) {
  int i = blockIdx.x * 256 + threadIdx.x;
  if (i >= N_NODES * OUT_F) return;
  float rs = rowsum[i >> 7];
  rs = (rs == 0.0f) ? 1.0f : rs;
  float v = out[i] / rs;
  out[i] = v > 0.0f ? v : (__expf(v) - 1.0f);
}

// ---------------------------------------------------------------------------
extern "C" void kernel_launch(void* const* d_in, const int* in_sizes, int n_in,
                              void* d_out, int out_size, void* d_ws, size_t ws_size,
                              hipStream_t stream) {
  const float* x    = (const float*)d_in[0];
  const float* W    = (const float*)d_in[1];
  const float* a    = (const float*)d_in[2];
  const int*   edge = (const int*)d_in[3];
  float* out = (float*)d_out;

  char* ws = (char*)d_ws;
  size_t off = 0;
  auto alloc = [&](size_t bytes) -> char* {
    char* p = ws + off;
    off = (off + bytes + 255) & ~(size_t)255;
    return p;
  };
  float*    hbuf   = (float*)alloc((size_t)N_NODES * OUT_F * sizeof(float)); // 51.2 MB
  _Float16* Wp     = (_Float16*)alloc((size_t)8 * 8 * 32 * 16 * sizeof(_Float16)); // 64 KB
  float*    s1     = (float*)alloc((size_t)N_NODES * sizeof(float));
  float*    s2     = (float*)alloc((size_t)N_NODES * sizeof(float));
  float*    rowsum = (float*)alloc((size_t)N_NODES * sizeof(float));

  const int total = N_NODES * OUT_F;
  zero_kernel<<<(total + 255) / 256, 256, 0, stream>>>(out, rowsum, total, N_NODES);
  pack_w_kernel<<<8, 256, 0, stream>>>(W, Wp);
  gemm_wmma_kernel<<<N_NODES / 16, 256, 0, stream>>>(x, Wp, hbuf);
  node_score_kernel<<<N_NODES / 8, 256, 0, stream>>>(hbuf, a, s1, s2);
  edge_agg_kernel<<<4096, 256, 0, stream>>>(edge, hbuf, s1, s2, rowsum, out);
  finalize_kernel<<<(total + 255) / 256, 256, 0, stream>>>(out, rowsum);
}
